// Align_74234214744316
// MI455X (gfx1250) — matrix operand order repaired and checked by reading, not compiled, not run on hardware
//
#include <hip/hip_runtime.h>
#include <hip/hip_bf16.h>

// ---------------------------------------------------------------------------
// EDVR-style PCD alignment pipeline for MI455X (gfx1250, wave32, WMMA).
// All 64/128-in-channel 3x3 convs (incl. the deformable-conv einsum) run as
// implicit GEMM on v_wmma_f32_16x16x32_f16 with f16 operands staged in LDS
// and fp32 accumulation. Round 2: Cin templated (kills SALU division chains),
// branchless clamped-address B-tile builds, packed ds_store_b128 staging.
// ---------------------------------------------------------------------------

typedef _Float16 v8h  __attribute__((ext_vector_type(8)));
typedef _Float16 v16h __attribute__((ext_vector_type(16)));
typedef float    v8f  __attribute__((ext_vector_type(8)));

#define NF   64
#define DG   8
#define K2   9
#define H_   128
#define W_   128
#define HWSZ (H_ * W_)
#define BB   4
#define NN   5
#define BN_  (BB * NN)

static __device__ __forceinline__ v16h combine16(v8h lo, v8h hi) {
  v16h r;
#pragma unroll
  for (int i = 0; i < 8; ++i) { r[i] = lo[i]; r[i + 8] = hi[i]; }
  return r;
}

// ---------------------------------------------------------------------------
// Weight repack: fp32 [O][Cin][3][3] -> f16 [Mpad][Cin*9], k = tap*Cin + cin.
// ---------------------------------------------------------------------------
__global__ void pack_w_kernel(const float* __restrict__ w, _Float16* __restrict__ A,
                              int O, int Cin, int Mpad) {
  long idx = (long)blockIdx.x * 256 + threadIdx.x;
  int Kpad = Cin * 9;
  long tot = (long)Mpad * Kpad;
  if (idx >= tot) return;
  int m = (int)(idx / Kpad);
  int k = (int)(idx % Kpad);
  int tap = k / Cin, c = k % Cin;
  float v = (m < O) ? w[((long)m * Cin + c) * 9 + tap] : 0.0f;
  A[(long)m * Kpad + k] = (_Float16)v;
}

// ---------------------------------------------------------------------------
// Generic 3x3 SAME conv as implicit GEMM. CIN is compile-time (64 or 128).
// Block = 256 threads (8 waves). Tile: 64 Cout x 128 pixels (one image row).
// ---------------------------------------------------------------------------
template <int CIN>
__global__ __launch_bounds__(256)
void conv3x3_igemm(const float* __restrict__ X1, const float* __restrict__ X2,
                   long long imgStride1, long long imgStride2,
                   const _Float16* __restrict__ A, const float* __restrict__ bias,
                   const float* __restrict__ Res, long long resImgStride,
                   float* __restrict__ Y, long long yImgStride,
                   int Cout, int relu) {
  __shared__ __attribute__((aligned(16))) _Float16 lA[64][48];
  __shared__ __attribute__((aligned(16))) _Float16 lB[128][48];

  constexpr int KPAD = CIN * 9;
  constexpr int CSTEPS = CIN / 32;   // K chunks per tap (2 or 4)

  const int t = threadIdx.x;
  const int lane = t & 31, wave = t >> 5;
  const int y = blockIdx.x;           // image row
  const int m0 = blockIdx.y * 64;     // Cout tile
  const int img = blockIdx.z;

  const float* x1 = X1 + (long long)img * imgStride1;
  const float* x2 = X2 ? (X2 + (long long)img * imgStride2) : nullptr;

  // per-thread constant pieces of the staging work
  const int aRow = t >> 2;            // A staging row 0..63
  const int aCol = (t & 3) * 8;       // A staging col {0,8,16,24}
  const int bN = t & 127;             // B staging pixel
  const int bKh = t >> 7;             // 0/1 -> kk 0..15 / 16..31

  const int mw = (wave >> 1) * 16;
  const int nwv = (wave & 1) * 64;
  const int khalf = lane >> 4;
  const int l15 = lane & 15;

  v8f acc[4] = {v8f{}, v8f{}, v8f{}, v8f{}};

  const _Float16* aSrc = A + (long long)(m0 + aRow) * KPAD + aCol;

#pragma unroll 1
  for (int tap = 0; tap < 9; ++tap) {
    const int dy = tap / 3 - 1, dx = tap % 3 - 1;   // compile-friendly consts
    const int yy = y + dy;
    const int yyc = min(max(yy, 0), H_ - 1);
    const float ymask = ((unsigned)yy < (unsigned)H_) ? 1.0f : 0.0f;
    const int xx = bN + dx;
    const int xxc = min(max(xx, 0), W_ - 1);
    const float mask = (((unsigned)xx < (unsigned)W_) ? 1.0f : 0.0f) * ymask;
    const long long sp = (long long)yyc * W_ + xxc;   // always in-bounds

#pragma unroll 1
    for (int cq = 0; cq < CSTEPS; ++cq) {
      const int c0 = cq * 32;
      // ---- stage A tile (64x32 halves): one b128 per thread ----
      *(v8h*)&lA[aRow][aCol] = *(const v8h*)aSrc;
      aSrc += 32;
      __builtin_prefetch(aSrc, 0, 0);  // next A chunk -> global_prefetch_b8
      // ---- build B tile: branchless clamped loads, packed b128 stores ----
      {
        int cin0 = c0 + bKh * 16;
        v8h p0, p1;
#pragma unroll
        for (int j = 0; j < 8; ++j) {
          int cin = cin0 + j;
          const float* src = (x2 && cin >= 64) ? (x2 + (long long)(cin - 64) * HWSZ)
                                               : (x1 + (long long)cin * HWSZ);
          p0[j] = (_Float16)(src[sp] * mask);
        }
#pragma unroll
        for (int j = 0; j < 8; ++j) {
          int cin = cin0 + 8 + j;
          const float* src = (x2 && cin >= 64) ? (x2 + (long long)(cin - 64) * HWSZ)
                                               : (x1 + (long long)cin * HWSZ);
          p1[j] = (_Float16)(src[sp] * mask);
        }
        *(v8h*)&lB[bN][bKh * 16] = p0;
        *(v8h*)&lB[bN][bKh * 16 + 8] = p1;
      }
      __syncthreads();

      // ---- WMMA: wave -> 16(M) x 64(N) sub-tile ----
      {
        v16h a = combine16(*(const v8h*)&lA[mw + l15][khalf * 8],
                           *(const v8h*)&lA[mw + l15][khalf * 8 + 16]);
#pragma unroll
        for (int s = 0; s < 4; ++s) {
          int bRow = nwv + s * 16 + l15;
          v16h b = combine16(*(const v8h*)&lB[bRow][khalf * 8],
                             *(const v8h*)&lB[bRow][khalf * 8 + 16]);
          acc[s] = __builtin_amdgcn_wmma_f32_16x16x32_f16(
              false, a, false, b, (short)0, acc[s], false, false);
        }
      }
      __syncthreads();
    }
  }

  // ---- epilogue: bias (+ReLU) (+residual) -> fp32 store ----
  const int lhi = lane >> 4;
  float* yimg = Y + (long long)img * yImgStride;
  const float* rimg = Res ? (Res + (long long)img * resImgStride) : nullptr;
#pragma unroll
  for (int s = 0; s < 4; ++s) {
#pragma unroll
    for (int r = 0; r < 8; ++r) {
      int m = m0 + mw + r + 8 * lhi;
      int n = nwv + s * 16 + l15;
      if (m < Cout) {
        float v = acc[s][r] + bias[m];
        if (relu) v = fmaxf(v, 0.0f);
        long long o = (long long)m * HWSZ + (long long)y * W_ + n;
        if (rimg) v += rimg[o];
        yimg[o] = v;
      }
    }
  }
}

// ---------------------------------------------------------------------------
// Modulated deformable conv (DCNv2) main conv as implicit GEMM. B-tile is
// bilinear-gathered & mask-modulated; validity folded into bilinear weights
// (branchless), 8-channel groups stored as one ds_store_b128.
// OM layout: [img][216][H][W]; ch o1 = g*9+tap, o2 = +72, mask = +144.
// ---------------------------------------------------------------------------
__global__ __launch_bounds__(256)
void dcn_igemm(const float* __restrict__ X, long long xImgStride,
               const float* __restrict__ OM,
               const _Float16* __restrict__ A, const float* __restrict__ bias,
               float* __restrict__ Y, long long yImgStride) {
  __shared__ __attribute__((aligned(16))) _Float16 lA[64][48];
  __shared__ __attribute__((aligned(16))) _Float16 lB[128][48];

  constexpr int KPAD = NF * 9;  // 576

  const int t = threadIdx.x;
  const int lane = t & 31, wave = t >> 5;
  const int y = blockIdx.x;
  const int img = blockIdx.z;

  const float* x = X + (long long)img * xImgStride;
  const float* om = OM + (long long)img * (long long)(3 * DG * K2) * HWSZ;

  const int aRow = t >> 2;
  const int aCol = (t & 3) * 8;
  const int bN = t & 127;
  const int bKh = t >> 7;
  const long long ospat = (long long)y * W_ + bN;

  const int mw = (wave >> 1) * 16;
  const int nwv = (wave & 1) * 64;
  const int khalf = lane >> 4;
  const int l15 = lane & 15;

  v8f acc[4] = {v8f{}, v8f{}, v8f{}, v8f{}};

  const _Float16* aSrc = A + (long long)aRow * KPAD + aCol;

#pragma unroll 1
  for (int tap = 0; tap < 9; ++tap) {
    const float ky = (float)(tap / 3 - 1);
    const float kx = (float)(tap % 3 - 1);
#pragma unroll 1
    for (int cq = 0; cq < 2; ++cq) {
      const int c0 = cq * 32;
      // ---- A tile ----
      *(v8h*)&lA[aRow][aCol] = *(const v8h*)aSrc;
      aSrc += 32;
      // ---- B tile: deformable bilinear sampling, branchless ----
#pragma unroll
      for (int jg = 0; jg < 2; ++jg) {
        const int kk0 = bKh * 16 + jg * 8;
        const int cin0 = c0 + kk0;
        const int g = cin0 >> 3;
        const int ch = g * K2 + tap;
        float dyv = om[(long long)ch * HWSZ + ospat];
        float dxv = om[(long long)(ch + DG * K2) * HWSZ + ospat];
        float mv  = om[(long long)(ch + 2 * DG * K2) * HWSZ + ospat];
        mv = 1.0f / (1.0f + __expf(-mv));  // sigmoid -> v_exp_f32
        float py = (float)y + ky + dyv;
        float px = (float)bN + kx + dxv;
        float y0f = floorf(py), x0f = floorf(px);
        float wy = py - y0f, wx = px - x0f;
        int y0 = (int)y0f, x0 = (int)x0f;
        // validity folded into weights; loads use clamped (safe) addresses
        float my0 = ((unsigned)y0 < (unsigned)H_) ? 1.0f : 0.0f;
        float my1 = ((unsigned)(y0 + 1) < (unsigned)H_) ? 1.0f : 0.0f;
        float mx0 = ((unsigned)x0 < (unsigned)W_) ? 1.0f : 0.0f;
        float mx1 = ((unsigned)(x0 + 1) < (unsigned)W_) ? 1.0f : 0.0f;
        int y0c = min(max(y0, 0), H_ - 1), y1c = min(max(y0 + 1, 0), H_ - 1);
        int x0c = min(max(x0, 0), W_ - 1), x1c = min(max(x0 + 1, 0), W_ - 1);
        float w00 = (1.0f - wy) * (1.0f - wx) * mv * my0 * mx0;
        float w01 = (1.0f - wy) * wx * mv * my0 * mx1;
        float w10 = wy * (1.0f - wx) * mv * my1 * mx0;
        float w11 = wy * wx * mv * my1 * mx1;
        long long p00 = (long long)y0c * W_ + x0c, p01 = (long long)y0c * W_ + x1c;
        long long p10 = (long long)y1c * W_ + x0c, p11 = (long long)y1c * W_ + x1c;
        v8h pk;
#pragma unroll
        for (int c = 0; c < 8; ++c) {
          const float* xc = x + (long long)(cin0 + c) * HWSZ;
          pk[c] = (_Float16)(w00 * xc[p00] + w01 * xc[p01] +
                             w10 * xc[p10] + w11 * xc[p11]);
        }
        *(v8h*)&lB[bN][kk0] = pk;
      }
      __syncthreads();

      {
        v16h a = combine16(*(const v8h*)&lA[mw + l15][khalf * 8],
                           *(const v8h*)&lA[mw + l15][khalf * 8 + 16]);
#pragma unroll
        for (int s = 0; s < 4; ++s) {
          int bRow = nwv + s * 16 + l15;
          v16h b = combine16(*(const v8h*)&lB[bRow][khalf * 8],
                             *(const v8h*)&lB[bRow][khalf * 8 + 16]);
          acc[s] = __builtin_amdgcn_wmma_f32_16x16x32_f16(
              false, a, false, b, (short)0, acc[s], false, false);
        }
      }
      __syncthreads();
    }
  }

  const int lhi = lane >> 4;
  float* yimg = Y + (long long)img * yImgStride;
#pragma unroll
  for (int s = 0; s < 4; ++s) {
#pragma unroll
    for (int r = 0; r < 8; ++r) {
      int m = mw + r + 8 * lhi;
      int n = nwv + s * 16 + l15;
      yimg[(long long)m * HWSZ + (long long)y * W_ + n] = acc[s][r] + bias[m];
    }
  }
}

// ---------------------------------------------------------------------------
// Tiny layers on plain VALU (negligible FLOPs): Cin=1 init conv, Cout=1 rec.
// ---------------------------------------------------------------------------
__global__ void conv_init_kernel(const float* __restrict__ X, const float* __restrict__ Wt,
                                 const float* __restrict__ Bi, float* __restrict__ Y) {
  long idx = (long)blockIdx.x * 256 + threadIdx.x;
  if (idx >= (long)BN_ * NF * HWSZ) return;
  int p = (int)(idx % HWSZ);
  int oc = (int)((idx / HWSZ) % NF);
  int img = (int)(idx / ((long)NF * HWSZ));
  int yy = p / W_, xx = p % W_;
  const float* xi = X + (long)img * HWSZ;
  float s = Bi[oc];
#pragma unroll
  for (int tap = 0; tap < 9; ++tap) {
    int iy = yy + tap / 3 - 1, ix = xx + tap % 3 - 1;
    if ((unsigned)iy < (unsigned)H_ && (unsigned)ix < (unsigned)W_)
      s += Wt[oc * 9 + tap] * xi[iy * W_ + ix];
  }
  Y[idx] = fmaxf(s, 0.0f);
}

__global__ void conv_rec_kernel(const float* __restrict__ F, const float* __restrict__ Wt,
                                const float* __restrict__ Bi, float* __restrict__ Out,
                                int neigh) {
  long idx = (long)blockIdx.x * 256 + threadIdx.x;
  if (idx >= (long)BB * HWSZ) return;
  int p = (int)(idx % HWSZ);
  int b = (int)(idx / HWSZ);
  int yy = p / W_, xx = p % W_;
  const float* f = F + (long)b * NF * HWSZ;
  float s = Bi[0];
  for (int c = 0; c < NF; ++c) {
    const float* fc = f + (long)c * HWSZ;
#pragma unroll
    for (int tap = 0; tap < 9; ++tap) {
      int iy = yy + tap / 3 - 1, ix = xx + tap % 3 - 1;
      if ((unsigned)iy < (unsigned)H_ && (unsigned)ix < (unsigned)W_)
        s += Wt[c * 9 + tap] * fc[iy * W_ + ix];
    }
  }
  Out[((long)b * NN + neigh) * HWSZ + p] = s;
}

// ---------------------------------------------------------------------------
// Orchestration
// ---------------------------------------------------------------------------
extern "C" void kernel_launch(void* const* d_in, const int* in_sizes, int n_in,
                              void* d_out, int out_size, void* d_ws, size_t ws_size,
                              hipStream_t stream) {
  const float* x      = (const float*)d_in[0];
  const float* w_init = (const float*)d_in[1];
  const float* b_init = (const float*)d_in[2];
  const float* res_w1 = (const float*)d_in[3];
  const float* res_b1 = (const float*)d_in[4];
  const float* res_w2 = (const float*)d_in[5];
  const float* res_b2 = (const float*)d_in[6];
  const float* w_bn   = (const float*)d_in[7];
  const float* b_bn   = (const float*)d_in[8];
  const float* off_w  = (const float*)d_in[9];
  const float* off_b  = (const float*)d_in[10];
  const float* com_w  = (const float*)d_in[11];
  const float* com_b  = (const float*)d_in[12];
  const float* dcn_w  = (const float*)d_in[13];
  const float* dcn_b  = (const float*)d_in[14];
  const float* w_rec  = (const float*)d_in[15];
  const float* b_rec  = (const float*)d_in[16];
  float* out = (float*)d_out;

  // --- workspace carve-up (~265 MB used) ---
  char* wsp = (char*)d_ws;
  auto alloc = [&](size_t bytes) {
    char* p = wsp;
    wsp += (bytes + 255) & ~(size_t)255;
    return p;
  };
  float* feat = (float*)alloc((size_t)BN_ * NF * HWSZ * 4);         // 80 MB
  float* htmp = (float*)alloc((size_t)BN_ * NF * HWSZ * 4);         // 80 MB
  float* fea  = (float*)alloc((size_t)BB * NF * HWSZ * 4);          // 16 MB
  float* obuf = (float*)alloc((size_t)BB * NF * HWSZ * 4);          // 16 MB
  float* ombf = (float*)alloc((size_t)BB * 3 * DG * K2 * HWSZ * 4); // 54 MB
  float* fea2 = (float*)alloc((size_t)BB * NF * HWSZ * 4);          // 16 MB
  const int K64 = NF * 9, K128 = 2 * NF * 9, MCOM = 256;
  _Float16* pkRes1 = (_Float16*)alloc((size_t)5 * NF * K64 * 2);
  _Float16* pkRes2 = (_Float16*)alloc((size_t)5 * NF * K64 * 2);
  _Float16* pkBn   = (_Float16*)alloc((size_t)NF * K128 * 2);
  _Float16* pkOff  = (_Float16*)alloc((size_t)4 * NF * K64 * 2);
  _Float16* pkCom  = (_Float16*)alloc((size_t)4 * MCOM * K64 * 2);
  _Float16* pkDcn  = (_Float16*)alloc((size_t)4 * NF * K64 * 2);

  auto pack = [&](const float* w, _Float16* A, int O, int Cin, int Mpad) {
    long tot = (long)Mpad * Cin * 9;
    pack_w_kernel<<<(unsigned)((tot + 255) / 256), 256, 0, stream>>>(w, A, O, Cin, Mpad);
  };
  for (int i = 0; i < 5; ++i) {
    pack(res_w1 + (long)i * NF * NF * 9, pkRes1 + (long)i * NF * K64, NF, NF, NF);
    pack(res_w2 + (long)i * NF * NF * 9, pkRes2 + (long)i * NF * K64, NF, NF, NF);
  }
  pack(w_bn, pkBn, NF, 2 * NF, NF);
  for (int j = 0; j < 4; ++j) {
    pack(off_w + (long)j * NF * NF * 9, pkOff + (long)j * NF * K64, NF, NF, NF);
    pack(com_w + (long)j * (3 * DG * K2) * NF * 9, pkCom + (long)j * MCOM * K64,
         3 * DG * K2, NF, MCOM);
    pack(dcn_w + (long)j * NF * NF * 9, pkDcn + (long)j * NF * K64, NF, NF, NF);
  }

  const long long fstr = (long long)NF * HWSZ;     // per-image 64ch stride
  const long long nstr = (long long)NN * NF * HWSZ;

  // init conv (Cin=1) + ReLU
  {
    long tot = (long)BN_ * NF * HWSZ;
    conv_init_kernel<<<(unsigned)((tot + 255) / 256), 256, 0, stream>>>(x, w_init, b_init, feat);
  }
  // residual blocks: h = relu(conv(out)); out += conv(h)
  for (int i = 0; i < 5; ++i) {
    conv3x3_igemm<64><<<dim3(H_, 1, BN_), 256, 0, stream>>>(
        feat, nullptr, fstr, 0, pkRes1 + (long)i * NF * K64, res_b1 + i * NF,
        nullptr, 0, htmp, fstr, NF, 1);
    conv3x3_igemm<64><<<dim3(H_, 1, BN_), 256, 0, stream>>>(
        htmp, nullptr, fstr, 0, pkRes2 + (long)i * NF * K64, res_b2 + i * NF,
        feat, fstr, feat, fstr, NF, 0);
  }

  // per-neighbor alignment
  for (int i = 0; i < 5; ++i) {
    const float* refp = feat + 2LL * NF * HWSZ;    // frame N//2, stride nstr
    const float* neip = feat + (long long)i * NF * HWSZ;
    // fea = conv(concat(ref, nei), w_bn)
    conv3x3_igemm<128><<<dim3(H_, 1, BB), 256, 0, stream>>>(
        refp, neip, nstr, nstr, pkBn, b_bn, nullptr, 0, fea, fstr, NF, 0);
    float* cur = fea;
    float* nxt = fea2;
    for (int j = 0; j < 4; ++j) {
      // o = conv(cur, off_w[j])
      conv3x3_igemm<64><<<dim3(H_, 1, BB), 256, 0, stream>>>(
          cur, nullptr, fstr, 0, pkOff + (long)j * NF * K64, off_b + j * NF,
          nullptr, 0, obuf, fstr, NF, 0);
      // om = conv(o, com_w[j])  (Cout=216, M tiles of 64)
      conv3x3_igemm<64><<<dim3(H_, 4, BB), 256, 0, stream>>>(
          obuf, nullptr, fstr, 0, pkCom + (long)j * MCOM * K64, com_b + j * 3 * DG * K2,
          nullptr, 0, ombf, (long long)(3 * DG * K2) * HWSZ, 3 * DG * K2, 0);
      // nxt = dcn(x_src, om, dcn_w[j])  (j==2 samples the raw neighbor)
      const float* xsrc = (j == 2) ? neip : cur;
      long long xstr = (j == 2) ? nstr : fstr;
      dcn_igemm<<<dim3(H_, 1, BB), 256, 0, stream>>>(
          xsrc, xstr, ombf, pkDcn + (long)j * NF * K64, dcn_b + j * NF, nxt, fstr);
      float* tmp = cur; cur = nxt; nxt = tmp;
    }
    // reconstruction conv (Cout=1) into output slot [b][i]
    {
      long tot = (long)BB * HWSZ;
      conv_rec_kernel<<<(unsigned)((tot + 255) / 256), 256, 0, stream>>>(
          cur, w_rec, b_rec, out, i);
    }
  }
}
